// nsctnet_25984552141026
// MI455X (gfx1250) — compile-verified, matrix operand order checked by reading, not run on hardware
//
#include <hip/hip_runtime.h>
#include <hip/hip_bf16.h>

// ---------------------------------------------------------------------------
// Types / helpers
// ---------------------------------------------------------------------------
typedef __attribute__((ext_vector_type(8)))  float          v8f;
typedef __attribute__((ext_vector_type(16))) __bf16         v16bf;
typedef __attribute__((ext_vector_type(16))) unsigned short v16u;

#define TOKENS   100352L   // 8*112*112
#define BDIM     160
#define HH       112
#define WW2      112
#define BB       8
#define CC       16

__device__ __forceinline__ unsigned short f2bf(float f) {
    unsigned int u = __builtin_bit_cast(unsigned int, f);
    unsigned int r = u + 0x7fffu + ((u >> 16) & 1u);   // round-to-nearest-even
    return (unsigned short)(r >> 16);
}
__device__ __forceinline__ float gelu_tanh(float x) {
    float x3 = x * x * x;
    float t  = tanhf(0.7978845608028654f * (x + 0.044715f * x3));
    return 0.5f * x * (1.0f + t);
}

// ---------------------------------------------------------------------------
// fp32 -> bf16 convert (dense)
// ---------------------------------------------------------------------------
__global__ void k_convert_bf16(const float* __restrict__ s,
                               unsigned short* __restrict__ d, long n) {
    long i = (long)blockIdx.x * 256 + threadIdx.x;
    if (i < n) d[i] = f2bf(s[i]);
}

// fp32 [rows,N] -> bf16 [rows,Npad] with zero padding (Npad % 64 == 0)
__global__ void k_convert_pad(const float* __restrict__ s,
                              unsigned short* __restrict__ d,
                              long rows, int N, int Npad) {
    long i = (long)blockIdx.x * 256 + threadIdx.x;
    if (i >= rows * Npad) return;
    long k = i / Npad; int n = (int)(i % Npad);
    d[i] = (n < N) ? f2bf(s[k * N + n]) : (unsigned short)0;
}

// ---------------------------------------------------------------------------
// relu + 9-subband 9x9 analysis filter bank + concat -> x3 NHWC [T,160]
// channel order: k*16 + c for k in 0..8, then skip channels 144+c = relu(x)
// ---------------------------------------------------------------------------
__global__ __launch_bounds__(256) void k_relu_ct(const float* __restrict__ x,
                                                 const float* __restrict__ fa,
                                                 float* __restrict__ x3) {
    __shared__ float fs[9 * 81];
    for (int i = threadIdx.x; i < 729; i += 256) fs[i] = fa[i];
    __syncthreads();
    long idx = (long)blockIdx.x * 256 + threadIdx.x;      // over B*C*H*W
    if (idx >= (long)BB * CC * HH * WW2) return;
    int w = idx % WW2;  long r = idx / WW2;
    int h = r % HH;     r /= HH;
    int c = r % CC;     int b = (int)(r / CC);
    const float* xp = x + ((long)(b * CC + c)) * HH * WW2;
    float acc[9];
#pragma unroll
    for (int k = 0; k < 9; ++k) acc[k] = 0.f;
    float center = 0.f;
    for (int fh = 0; fh < 9; ++fh) {
        int hh = h + fh - 4;
        if (hh < 0 || hh >= HH) continue;
        for (int fw = 0; fw < 9; ++fw) {
            int ww = w + fw - 4;
            if (ww < 0 || ww >= WW2) continue;
            float v = xp[hh * WW2 + ww];
            v = v > 0.f ? v : 0.f;
            if (fh == 4 && fw == 4) center = v;
#pragma unroll
            for (int k = 0; k < 9; ++k) acc[k] += v * fs[k * 81 + fh * 9 + fw];
        }
    }
    long tbase = (((long)b * HH + h) * WW2 + w) * BDIM;
#pragma unroll
    for (int k = 0; k < 9; ++k) x3[tbase + k * CC + c] = acc[k];
    x3[tbase + 144 + c] = center;
}

// ---------------------------------------------------------------------------
// LayerNorm over C=160, one wave per token, bf16 output
// ---------------------------------------------------------------------------
__global__ __launch_bounds__(256) void k_layernorm(const float* __restrict__ x,
                                                   const float* __restrict__ g,
                                                   const float* __restrict__ bt,
                                                   unsigned short* __restrict__ out,
                                                   long T) {
    long t = (long)blockIdx.x * 8 + (threadIdx.x >> 5);
    int lane = threadIdx.x & 31;
    if (t >= T) return;
    const float* p = x + t * BDIM;
    float v[5]; float s = 0.f;
#pragma unroll
    for (int i = 0; i < 5; ++i) { v[i] = p[lane + i * 32]; s += v[i]; }
#pragma unroll
    for (int m = 16; m; m >>= 1) s += __shfl_xor(s, m, 32);
    float mu = s * (1.f / 160.f);
    float q = 0.f;
#pragma unroll
    for (int i = 0; i < 5; ++i) { float d = v[i] - mu; q += d * d; }
#pragma unroll
    for (int m = 16; m; m >>= 1) q += __shfl_xor(q, m, 32);
    float inv = rsqrtf(q * (1.f / 160.f) + 1e-5f);
#pragma unroll
    for (int i = 0; i < 5; ++i) {
        int c = lane + i * 32;
        out[t * BDIM + c] = f2bf((v[i] - mu) * inv * g[c] + bt[c]);
    }
}

// ---------------------------------------------------------------------------
// bf16 WMMA GEMM: Out[M,N] = A[M,K] * B[K,ldb](zero-padded to ldb) + bias
//   flags: 1 = bf16 output, 2 = gelu; residual != nullptr -> fused add (fp32)
//   M % 128 == 0, K % 32 == 0, ldb % 64 == 0 (B pre-padded -> no load guards)
// Block 256 (8 waves); tile 128x64x32; wave = 16 rows x 64 cols (4 WMMA accs)
// ---------------------------------------------------------------------------
#define GP 40   // LDS ushort row stride (80B, 16B aligned)
__global__ __launch_bounds__(256) void k_gemm_bf16(
        const unsigned short* __restrict__ A, const unsigned short* __restrict__ Bm,
        const float* __restrict__ bias, const float* __restrict__ residual,
        void* __restrict__ Out, int M, int N, int ldb, int K, int flags) {
    __shared__ __align__(16) unsigned short As[128][GP];
    __shared__ __align__(16) unsigned short Bs[64][GP];   // transposed: [n][k]
    int tid  = threadIdx.x;
    int wid  = tid >> 5, lane = tid & 31;
    int half = lane >> 4, l15 = lane & 15;
    int bm = blockIdx.y * 128, bn = blockIdx.x * 64;
    int mt = wid;                                   // 0..7 (16-row m-tile)
    v8f acc[4];
#pragma unroll
    for (int t = 0; t < 4; ++t) acc[t] = (v8f){0.f,0.f,0.f,0.f,0.f,0.f,0.f,0.f};

    for (int kt = 0; kt < K; kt += 32) {
        __syncthreads();
        // ---- A tile 128x32: two 16B loads per thread, vector LDS stores
#pragma unroll
        for (int l = 0; l < 2; ++l) {
            int lin = tid + l * 256;
            int r = lin >> 2, c = (lin & 3) * 8;
            const unsigned short* pa = A + (long)(bm + r) * K + kt + c;
            *(uint4*)&As[r][c] = *(const uint4*)pa;
            __builtin_prefetch(pa + 32, 0, 1);   // speculative: safe past end
        }
        // ---- B tile 32x64: one 16B load per thread, transposed scatter store
        {
            int kr = tid >> 3, n0 = (tid & 7) * 8;
            const unsigned short* pb = Bm + (long)(kt + kr) * ldb + bn + n0;
            unsigned short e[8];
            *(uint4*)e = *(const uint4*)pb;
            __builtin_prefetch(pb + 32 * ldb, 0, 1);
#pragma unroll
            for (int i = 0; i < 8; ++i) Bs[n0 + i][kr] = e[i];
        }
        __syncthreads();
        // ---- fragments: load A frag + ALL 4 B frags, then WMMAs back-to-back
        v16u au, bu[4];
        {
            const unsigned short* p = &As[mt * 16 + l15][half * 8];
#pragma unroll
            for (int i = 0; i < 8; ++i) { au[i] = p[i]; au[8 + i] = p[16 + i]; }
        }
#pragma unroll
        for (int t = 0; t < 4; ++t) {
            const unsigned short* p = &Bs[t * 16 + l15][half * 8];
#pragma unroll
            for (int i = 0; i < 8; ++i) { bu[t][i] = p[i]; bu[t][8 + i] = p[16 + i]; }
        }
#pragma unroll
        for (int t = 0; t < 4; ++t)
            acc[t] = __builtin_amdgcn_wmma_f32_16x16x32_bf16(
                         false, __builtin_bit_cast(v16bf, au),
                         false, __builtin_bit_cast(v16bf, bu[t]),
                         (short)0, acc[t], false, false);
    }
    bool obf = flags & 1, gel = flags & 2;
#pragma unroll
    for (int t = 0; t < 4; ++t) {
        int col = bn + t * 16 + l15;
        if (col < N) {
            float bi = bias ? bias[col] : 0.f;
#pragma unroll
            for (int i = 0; i < 8; ++i) {
                long row = bm + mt * 16 + half * 8 + i;
                float v = acc[t][i] + bi;
                if (residual) v += residual[row * N + col];
                if (gel) v = gelu_tanh(v);
                if (obf) ((unsigned short*)Out)[row * N + col] = f2bf(v);
                else     ((float*)Out)[row * N + col] = v;
            }
        }
    }
}

// ---------------------------------------------------------------------------
// Windowed attention, one workgroup per (window, head), 128 threads (4 waves)
// S = q k^T via 16 WMMA (K=32 = head_dim), softmax(+rpb bias, +shift mask),
// O = P v via 16 WMMA, scatter with roll reversal.
// ---------------------------------------------------------------------------
__global__ __launch_bounds__(128) void k_win_attn(
        const unsigned short* __restrict__ qkv, const float* __restrict__ rpb,
        unsigned short* __restrict__ attn_out, int shift, int blk) {
    __shared__ __align__(16) unsigned short qs[64][40];
    __shared__ __align__(16) unsigned short ks_[64][40];
    __shared__ __align__(16) unsigned short vt[32][72];    // [channel][j]
    __shared__ float          sb[64][65];
    __shared__ __align__(16) unsigned short pbuf[64][72];
    __shared__ float          rps[169];

    int tid  = threadIdx.x;
    int b    = blockIdx.x >> 8;
    int wij  = blockIdx.x & 255;
    int wi   = wij >> 4, wj = wij & 15;
    int head = blockIdx.y;

    auto tok = [&](int e) -> long {
        int hr = wi * 7 + e / 7 + shift; if (hr >= HH)  hr -= HH;
        int wr = wj * 7 + e % 7 + shift; if (wr >= WW2) wr -= WW2;
        return ((long)(b * HH + hr)) * WW2 + wr;
    };

    // ---- gather q,k (vectorized 16B loads, contiguous LDS stores)
#pragma unroll
    for (int l = 0; l < 2; ++l) {
        int idx = tid + l * 128;              // 256 groups = 64 rows x 4
        int r = idx >> 2, d8 = (idx & 3) * 8;
        uint4 qv = {0u,0u,0u,0u}, kv = {0u,0u,0u,0u};
        if (r < 49) {
            long t = tok(r);
            qv = *(const uint4*)(qkv + t * 480 + head * 32 + d8);
            kv = *(const uint4*)(qkv + t * 480 + 160 + head * 32 + d8);
        }
        *(uint4*)&qs[r][d8]  = qv;
        *(uint4*)&ks_[r][d8] = kv;
    }
    // ---- gather v (vector load, transposed scatter store)
#pragma unroll
    for (int l = 0; l < 2; ++l) {
        int idx = tid + l * 128;
        int j = idx >> 2, d8 = (idx & 3) * 8;
        unsigned short e[8] = {0,0,0,0,0,0,0,0};
        if (j < 49)
            *(uint4*)e = *(const uint4*)(qkv + tok(j) * 480 + 320 + head * 32 + d8);
#pragma unroll
        for (int i = 0; i < 8; ++i) vt[d8 + i][j] = e[i];
    }
    for (int idx = tid; idx < 169; idx += 128)
        rps[idx] = rpb[(long)(blk * 169 + idx) * 5 + head];
    __syncthreads();

    int wid = tid >> 5, lane = tid & 31, half = lane >> 4, l15 = lane & 15;
    {   // ---- S = q k^T  (wave wid owns m-tile wid); batch frags, then WMMAs
        int mt = wid;
        v16u au, bu[4];
        const unsigned short* pa = &qs[mt * 16 + l15][half * 8];
#pragma unroll
        for (int i = 0; i < 8; ++i) { au[i] = pa[i]; au[8 + i] = pa[16 + i]; }
#pragma unroll
        for (int nt = 0; nt < 4; ++nt) {
            const unsigned short* pk = &ks_[nt * 16 + l15][half * 8];
#pragma unroll
            for (int i = 0; i < 8; ++i) { bu[nt][i] = pk[i]; bu[nt][8 + i] = pk[16 + i]; }
        }
        v8f sc[4];
#pragma unroll
        for (int nt = 0; nt < 4; ++nt) {
            sc[nt] = (v8f){0.f,0.f,0.f,0.f,0.f,0.f,0.f,0.f};
            sc[nt] = __builtin_amdgcn_wmma_f32_16x16x32_bf16(
                         false, __builtin_bit_cast(v16bf, au),
                         false, __builtin_bit_cast(v16bf, bu[nt]),
                         (short)0, sc[nt], false, false);
        }
#pragma unroll
        for (int nt = 0; nt < 4; ++nt)
#pragma unroll
            for (int i = 0; i < 8; ++i)
                sb[mt * 16 + half * 8 + i][nt * 16 + l15] = sc[nt][i];
    }
    __syncthreads();

    // ---- softmax rows (scale + rel-pos bias + shift mask), write P bf16
    const float scale = 0.17677669529663687f;   // 1/sqrt(32)
    for (int r = tid; r < 64; r += 128) {
        if (r < 49) {
            int rih = r / 7, riw = r % 7;
            int crr = 0;
            if (shift) {
                int hr = wi * 7 + rih, wr = wj * 7 + riw;
                int rh = hr < 105 ? 0 : (hr < 109 ? 1 : 2);
                int rw = wr < 105 ? 0 : (wr < 109 ? 1 : 2);
                crr = rh * 3 + rw;
            }
            float mx = -1e30f;
            for (int j = 0; j < 49; ++j) {
                int jih = j / 7, jiw = j % 7;
                float v = sb[r][j] * scale
                        + rps[(rih - jih + 6) * 13 + (riw - jiw + 6)];
                if (shift) {
                    int hr = wi * 7 + jih, wr = wj * 7 + jiw;
                    int rh = hr < 105 ? 0 : (hr < 109 ? 1 : 2);
                    int rw = wr < 105 ? 0 : (wr < 109 ? 1 : 2);
                    if (rh * 3 + rw != crr) v -= 100.f;
                }
                sb[r][j] = v;
                mx = v > mx ? v : mx;
            }
            float s = 0.f;
            for (int j = 0; j < 49; ++j) {
                float e = __expf(sb[r][j] - mx);
                sb[r][j] = e;
                s += e;
            }
            float inv = 1.f / s;
            for (int j = 0; j < 49; ++j) pbuf[r][j] = f2bf(sb[r][j] * inv);
            for (int j = 49; j < 64; ++j) pbuf[r][j] = 0;
        } else {
            for (int j = 0; j < 64; ++j) pbuf[r][j] = 0;
        }
    }
    __syncthreads();

    {   // ---- O = P v  (K = 64 -> 2 steps); batch frags per acc
        int mt = wid;
#pragma unroll
        for (int ntv = 0; ntv < 2; ++ntv) {
            v16u au[2], bu[2];
#pragma unroll
            for (int ks2 = 0; ks2 < 2; ++ks2) {
                const unsigned short* pa = &pbuf[mt * 16 + l15][ks2 * 32 + half * 8];
                const unsigned short* pv = &vt[ntv * 16 + l15][ks2 * 32 + half * 8];
#pragma unroll
                for (int i = 0; i < 8; ++i) {
                    au[ks2][i] = pa[i]; au[ks2][8 + i] = pa[16 + i];
                    bu[ks2][i] = pv[i]; bu[ks2][8 + i] = pv[16 + i];
                }
            }
            v8f c = {0.f,0.f,0.f,0.f,0.f,0.f,0.f,0.f};
#pragma unroll
            for (int ks2 = 0; ks2 < 2; ++ks2)
                c = __builtin_amdgcn_wmma_f32_16x16x32_bf16(
                        false, __builtin_bit_cast(v16bf, au[ks2]),
                        false, __builtin_bit_cast(v16bf, bu[ks2]),
                        (short)0, c, false, false);
#pragma unroll
            for (int i = 0; i < 8; ++i) {
                int row = mt * 16 + half * 8 + i;
                if (row < 49) {
                    long t = tok(row);
                    attn_out[t * BDIM + head * 32 + ntv * 16 + l15] = f2bf(c[i]);
                }
            }
        }
    }
}

// ---------------------------------------------------------------------------
// Synthesis filter bank: y[T,144] NHWC -> y_trans [B,16,H,W]
// ---------------------------------------------------------------------------
__global__ __launch_bounds__(256) void k_ict(const float* __restrict__ y,
                                             const float* __restrict__ fsyn,
                                             float* __restrict__ yt) {
    __shared__ float fs[729];
    for (int i = threadIdx.x; i < 729; i += 256) fs[i] = fsyn[i];
    __syncthreads();
    long idx = (long)blockIdx.x * 256 + threadIdx.x;
    if (idx >= (long)BB * CC * HH * WW2) return;
    int w = idx % WW2;  long r = idx / WW2;
    int h = r % HH;     r /= HH;
    int c = r % CC;     int b = (int)(r / CC);
    float acc = 0.f;
    for (int fh = 0; fh < 9; ++fh) {
        int hh = h + fh - 4;
        if (hh < 0 || hh >= HH) continue;
        for (int fw = 0; fw < 9; ++fw) {
            int ww = w + fw - 4;
            if (ww < 0 || ww >= WW2) continue;
            long tb = (((long)b * HH + hh) * WW2 + ww) * 144 + c;
#pragma unroll
            for (int k = 0; k < 9; ++k)
                acc += y[tb + k * CC] * fs[k * 81 + fh * 9 + fw];
        }
    }
    yt[idx] = acc;   // NCHW linear order matches idx decomposition
}

// ---------------------------------------------------------------------------
// Final 1x1 conv over concat([y_trans(16), y(144)]) -> out [B,16,H,W]
// ---------------------------------------------------------------------------
__global__ __launch_bounds__(256) void k_final(const float* __restrict__ yt,
                                               const float* __restrict__ y,
                                               const float* __restrict__ cw,
                                               const float* __restrict__ cb,
                                               float* __restrict__ out) {
    __shared__ float wsm[16 * BDIM];
    __shared__ float bsm[16];
    for (int i = threadIdx.x; i < 16 * BDIM; i += 256) wsm[i] = cw[i];
    if (threadIdx.x < 16) bsm[threadIdx.x] = cb[threadIdx.x];
    __syncthreads();
    long idx = (long)blockIdx.x * 256 + threadIdx.x;     // over B*H*W
    if (idx >= TOKENS) return;
    int w = idx % WW2;  long r = idx / WW2;
    int h = r % HH;     int b = (int)(r / HH);
    float in[BDIM];
#pragma unroll
    for (int d = 0; d < 16; ++d)
        in[d] = yt[((long)(b * CC + d) * HH + h) * WW2 + w];
    const float* yp = y + idx * 144;
#pragma unroll
    for (int d = 0; d < 144; ++d) in[16 + d] = yp[d];
    for (int c = 0; c < 16; ++c) {
        float a = bsm[c];
#pragma unroll
        for (int d = 0; d < BDIM; ++d) a += in[d] * wsm[c * BDIM + d];
        out[((long)(b * CC + c) * HH + h) * WW2 + w] = a;
    }
}

// ---------------------------------------------------------------------------
// Host orchestration
// ---------------------------------------------------------------------------
extern "C" void kernel_launch(void* const* d_in, const int* in_sizes, int n_in,
                              void* d_out, int out_size, void* d_ws, size_t ws_size,
                              hipStream_t stream) {
    const float* x     = (const float*)d_in[0];
    const float* f_ana = (const float*)d_in[1];
    const float* f_syn = (const float*)d_in[2];
    const float* ln1g  = (const float*)d_in[3];
    const float* ln1b  = (const float*)d_in[4];
    const float* qkvw  = (const float*)d_in[5];
    const float* qkvb  = (const float*)d_in[6];
    const float* projw = (const float*)d_in[7];
    const float* projb = (const float*)d_in[8];
    const float* rpb   = (const float*)d_in[9];
    const float* ln2g  = (const float*)d_in[10];
    const float* ln2b  = (const float*)d_in[11];
    const float* w1    = (const float*)d_in[12];
    const float* b1    = (const float*)d_in[13];
    const float* w2    = (const float*)d_in[14];
    const float* b2    = (const float*)d_in[15];
    const float* linw  = (const float*)d_in[16];
    const float* linb  = (const float*)d_in[17];
    const float* convw = (const float*)d_in[18];
    const float* convb = (const float*)d_in[19];
    float* out = (float*)d_out;

    const long T = TOKENS;
    char* p = (char*)d_ws;
    auto take = [&](size_t bytes) { char* r = p; p += (bytes + 255) & ~size_t(255); return r; };

    float*          x3    = (float*)         take(T * 160 * 4);
    unsigned short* xln   = (unsigned short*)take(T * 160 * 2);
    unsigned short* attn  = (unsigned short*)take(T * 160 * 2);
    unsigned short* u480  = (unsigned short*)take(T * 640 * 2);  // qkv then mlp-hidden
    float*          ybuf  = (float*)         take(T * 144 * 4);
    float*          ytr   = (float*)         take((long)BB * CC * HH * WW2 * 4);
    // padded bf16 weights (ldb multiple of 64)
    unsigned short* wq    = (unsigned short*)take(2L * 160 * 512 * 2);
    unsigned short* wp    = (unsigned short*)take(2L * 160 * 192 * 2);
    unsigned short* w1b   = (unsigned short*)take(2L * 160 * 640 * 2);
    unsigned short* w2b   = (unsigned short*)take(2L * 640 * 192 * 2);
    unsigned short* wl    = (unsigned short*)take(160L * 192 * 2);

    auto convpad = [&](const float* s, unsigned short* d, long rows, int N, int Npad) {
        long n = rows * Npad;
        k_convert_pad<<<(int)((n + 255) / 256), 256, 0, stream>>>(s, d, rows, N, Npad);
    };
    convpad(qkvw, wq, 2L * 160, 480, 512);
    convpad(projw, wp, 2L * 160, 160, 192);
    convpad(w1,  w1b, 2L * 160, 640, 640);
    convpad(w2,  w2b, 2L * 640, 160, 192);
    convpad(linw, wl, 160L,     144, 192);

    int npix = (int)((long)BB * CC * HH * WW2 / 256);   // 6272
    k_relu_ct<<<npix, 256, 0, stream>>>(x, f_ana, x3);

    auto gemm = [&](const unsigned short* A, const unsigned short* B,
                    const float* bias, const float* res, void* O,
                    int N, int ldb, int K, int flags) {
        dim3 grid((N + 63) / 64, (int)(T / 128));
        k_gemm_bf16<<<grid, 256, 0, stream>>>(A, B, bias, res, O,
                                              (int)T, N, ldb, K, flags);
    };

    for (int blkI = 0; blkI < 2; ++blkI) {
        int shift = blkI ? 3 : 0;
        // LN1 -> bf16 tokens
        k_layernorm<<<(int)(T / 8), 256, 0, stream>>>(x3, ln1g + blkI * 160,
                                                      ln1b + blkI * 160, xln, T);
        // QKV: [T,160] x [160,480] -> bf16
        gemm(xln, wq + (long)blkI * 160 * 512, qkvb + blkI * 480, nullptr,
             u480, 480, 512, 160, /*bf16*/1);
        // windowed attention
        k_win_attn<<<dim3(2048, 5), 128, 0, stream>>>(u480, rpb, attn, shift, blkI);
        // proj + residual: [T,160] x [160,160] + x3 -> x3 (fp32)
        gemm(attn, wp + (long)blkI * 160 * 192, projb + blkI * 160, x3,
             x3, 160, 192, 160, 0);
        // LN2
        k_layernorm<<<(int)(T / 8), 256, 0, stream>>>(x3, ln2g + blkI * 160,
                                                      ln2b + blkI * 160, xln, T);
        // MLP1 + gelu -> bf16 hidden [T,640]
        gemm(xln, w1b + (long)blkI * 160 * 640, b1 + blkI * 640, nullptr,
             u480, 640, 640, 160, /*bf16|gelu*/3);
        // MLP2 + residual -> x3
        gemm(u480, w2b + (long)blkI * 640 * 192, b2 + blkI * 160, x3,
             x3, 160, 192, 640, 0);
    }

    // final linear 160 -> 144 (no LN): convert x3 to bf16, GEMM -> ybuf fp32
    {
        long n = T * 160;
        k_convert_bf16<<<(int)((n + 255) / 256), 256, 0, stream>>>(x3, xln, n);
    }
    gemm(xln, wl, linb, nullptr, ybuf, 144, 192, 160, 0);

    // synthesis filter bank and fused final conv
    k_ict<<<npix, 256, 0, stream>>>(ybuf, f_syn, ytr);
    k_final<<<(int)((T + 255) / 256), 256, 0, stream>>>(ytr, ybuf, convw, convb, out);
}
